// StateSpaceLayer_27401891348814
// MI455X (gfx1250) — matrix operand order repaired and checked by reading, not compiled, not run on hardware
//
#include <hip/hip_runtime.h>
#include <cstdint>

#define BATCH  4
#define SEQ    4096
#define HID    1024
#define NST    256
#define TROWS  (BATCH*SEQ)   // 16384
#define CHUNKS 32
#define CLEN   (SEQ/CHUNKS)  // 128

typedef __attribute__((ext_vector_type(16))) __bf16 v16bf;
typedef __attribute__((ext_vector_type(8)))  float  v8f;

union FragB { v16bf v; uint4 q[2]; };

__device__ __forceinline__ unsigned short f32_to_bf16(float f) {
  unsigned int u = __float_as_uint(f);
  u += 0x7FFFu + ((u >> 16) & 1u);   // round-to-nearest-even
  return (unsigned short)(u >> 16);
}

__device__ __forceinline__ v8f v8f_zero() {
  v8f z;
  #pragma unroll
  for (int i = 0; i < 8; ++i) z[i] = 0.0f;
  return z;
}

// ---------------------------------------------------------------------------
// Prep: WB[m][h] = sum_n Bm[m][n]*W_in[n][h]  (bf16)   bB[m] = Bm[m]·b_in
// ---------------------------------------------------------------------------
__global__ __launch_bounds__(256) void prep_wb_kernel(
    const float* __restrict__ Bm, const float* __restrict__ W_in,
    const float* __restrict__ b_in, unsigned short* __restrict__ WBb,
    float* __restrict__ bB)
{
  __shared__ float brow[NST];
  const int m = blockIdx.x;
  for (int n = threadIdx.x; n < NST; n += 256) brow[n] = Bm[m*NST + n];
  __syncthreads();
  for (int h = threadIdx.x; h < HID; h += 256) {
    float s = 0.f;
    for (int n = 0; n < NST; ++n) s += brow[n] * W_in[n*HID + h];
    WBb[(size_t)m*HID + h] = f32_to_bf16(s);
  }
  if (threadIdx.x == 0) {
    float s = 0.f;
    for (int n = 0; n < NST; ++n) s += brow[n] * b_in[n];
    bB[m] = s;
  }
}

// dst[j][k] = bf16(src[k][j])   (src: rows x cols, k=row, j=col)
__global__ __launch_bounds__(256) void transpose_to_bf16(
    const float* __restrict__ src, unsigned short* __restrict__ dst,
    int rows, int cols)
{
  int idx = blockIdx.x * 256 + threadIdx.x;
  if (idx < rows * cols) {
    int k = idx / cols, j = idx % cols;
    dst[(size_t)j*rows + k] = f32_to_bf16(src[(size_t)k*cols + j]);
  }
}

__global__ __launch_bounds__(256) void conv_bf16(
    const float* __restrict__ src, unsigned short* __restrict__ dst, int n)
{
  int i = blockIdx.x * 256 + threadIdx.x;
  if (i < n) dst[i] = f32_to_bf16(src[i]);
}

// Mout = Min @ Min   (256x256 f32)
__global__ __launch_bounds__(256) void matsq(
    const float* __restrict__ Min, float* __restrict__ Mout)
{
  __shared__ float row[NST];
  const int i = blockIdx.x;
  for (int k = threadIdx.x; k < NST; k += 256) row[k] = Min[i*NST + k];
  __syncthreads();
  const int j = threadIdx.x;
  float s = 0.f;
  for (int k = 0; k < NST; ++k) s += row[k] * Min[k*NST + j];
  Mout[i*NST + j] = s;
}

// ---------------------------------------------------------------------------
// bf16 WMMA GEMM: Out[t][j] = sum_k Asrc[t][k]*Bw[j][k] + bias[j]
// Asrc f32 [Trows][K] (b128-loaded, converted to bf16 on LDS store),
// Bw bf16 [Nout][K] row-major (natural B-fragment layout).
// block = 256 thr (8 waves, 2Mx4N), block tile 64x128, KT=32.
// Software-pipelined: fetch tile k+1 to regs -> barrier -> WMMA on tile k
// from LDS -> convert/store tile k+1 into the other LDS buffer.
// ---------------------------------------------------------------------------
__global__ __launch_bounds__(256) void gemm_bf16_kernel(
    const float* __restrict__ Asrc, const unsigned short* __restrict__ Bw,
    const float* __restrict__ bias, float* __restrict__ Out,
    int Trows, int K, int Nout)
{
  __shared__ __attribute__((aligned(16))) unsigned short ldsA[2][64][40];
  __shared__ __attribute__((aligned(16))) unsigned short ldsB[2][128][40];
  const int tid  = threadIdx.x;
  const int wave = tid >> 5, lane = tid & 31;
  const int wm = wave >> 2, wn = wave & 3;
  const int half = lane >> 4, ln = lane & 15;
  const int tBase = blockIdx.x * 64;
  const int jBase = blockIdx.y * 128;
  const int nkt = K >> 5;

  v8f acc[2][2];
  acc[0][0] = v8f_zero(); acc[0][1] = v8f_zero();
  acc[1][0] = v8f_zero(); acc[1][1] = v8f_zero();

  float4 ra[2];
  uint4  rb[2];

  auto fetch_regs = [&](int kt) {
    const int k0 = kt << 5;
    #pragma unroll
    for (int i = 0; i < 2; ++i) {              // A: 64 rows x 8 float4
      const int idx = tid + i*256;             // 0..511
      const int r = idx >> 3, c4 = (idx & 7) << 2;
      ra[i] = *(const float4*)&Asrc[(size_t)(tBase + r)*K + k0 + c4];
    }
    #pragma unroll
    for (int i = 0; i < 2; ++i) {              // B: 128 rows x 4 uint4
      const int idx = tid + i*256;
      const int r = idx >> 2, q = (idx & 3) << 3;
      rb[i] = *(const uint4*)&Bw[(size_t)(jBase + r)*K + k0 + q];
    }
  };
  auto store_lds = [&](int buf) {
    #pragma unroll
    for (int i = 0; i < 2; ++i) {
      const int idx = tid + i*256;
      const int r = idx >> 3, c4 = (idx & 7) << 2;
      uint2 pk;
      pk.x = (unsigned)f32_to_bf16(ra[i].x) | ((unsigned)f32_to_bf16(ra[i].y) << 16);
      pk.y = (unsigned)f32_to_bf16(ra[i].z) | ((unsigned)f32_to_bf16(ra[i].w) << 16);
      *(uint2*)&ldsA[buf][r][c4] = pk;
    }
    #pragma unroll
    for (int i = 0; i < 2; ++i) {
      const int idx = tid + i*256;
      const int r = idx >> 2, q = (idx & 3) << 3;
      *(uint4*)&ldsB[buf][r][q] = rb[i];
    }
  };

  fetch_regs(0);
  store_lds(0);
  for (int kt = 0; kt < nkt; ++kt) {
    const int buf = kt & 1;
    const bool more = (kt + 1 < nkt);
    if (more) fetch_regs(kt + 1);   // global loads in flight during compute
    __syncthreads();                // tile kt resident in ldsA/ldsB[buf]
    #pragma unroll
    for (int mi = 0; mi < 2; ++mi) {
      FragB a;
      const int row = wm*32 + mi*16 + ln;
      a.q[0] = *(const uint4*)&ldsA[buf][row][half*8];
      a.q[1] = *(const uint4*)&ldsA[buf][row][16 + half*8];
      #pragma unroll
      for (int ni = 0; ni < 2; ++ni) {
        FragB b;
        const int col = wn*32 + ni*16 + ln;
        b.q[0] = *(const uint4*)&ldsB[buf][col][half*8];
        b.q[1] = *(const uint4*)&ldsB[buf][col][16 + half*8];
        acc[mi][ni] = __builtin_amdgcn_wmma_f32_16x16x32_bf16(
            false, a.v, false, b.v, (short)0, acc[mi][ni], false, false);
      }
    }
    if (more) store_lds(buf ^ 1);
  }

  #pragma unroll
  for (int mi = 0; mi < 2; ++mi)
    #pragma unroll
    for (int ni = 0; ni < 2; ++ni) {
      const int col = jBase + wn*32 + ni*16 + ln;
      const float bv = bias[col];
      #pragma unroll
      for (int r = 0; r < 8; ++r) {
        const int row = tBase + wm*32 + mi*16 + r + half*8;
        Out[(size_t)row*Nout + col] = acc[mi][ni][r] + bv;
      }
    }
}

// ---------------------------------------------------------------------------
// Chunked scan, passes 1 (mode 0) and 3 (mode 1).
// 8 blocks x 16 waves. Block g owns rows R = g*16+m, R=(chunk<<2)|batch.
// State V[16][256] lives in LDS as bf16; A^T rows are register-resident
// B-fragments. Per step: acc = (mode0 ? uB_t : 0) + V @ A via 8 chained WMMA.
// mode0: hs[t] = acc, V = acc.   mode1: hs[t] += acc, V = acc.
// ---------------------------------------------------------------------------
__global__ __launch_bounds__(512) void scan_kernel(
    const unsigned short* __restrict__ At,   // [256][256] bf16, At[j][k]=A[k][j]
    float* __restrict__ hs,                  // [TROWS][256] in/out (uB -> hs)
    const float* __restrict__ Hbuf,          // [CHUNKS][4][256] carries (mode1)
    int mode)
{
  __shared__ __attribute__((aligned(16))) unsigned short Vb[16][264];
  const int tid  = threadIdx.x;
  const int w    = tid >> 5, lane = tid & 31;
  const int half = lane >> 4, ln = lane & 15;
  const int g    = blockIdx.x;

  FragB bf[8];
  {
    const unsigned short* base = At + ((size_t)(16*w + ln))*NST + half*8;
    #pragma unroll
    for (int kb = 0; kb < 8; ++kb) {
      bf[kb].q[0] = *(const uint4*)(base + kb*32);
      bf[kb].q[1] = *(const uint4*)(base + kb*32 + 16);
    }
  }

  for (int i = tid; i < 16*NST; i += 512) {
    int r = i >> 8, c = i & 255;
    float v = 0.f;
    if (mode == 1) {
      int R = g*16 + r, cch = R >> 2, b = R & 3;
      v = Hbuf[((size_t)cch*4 + b)*NST + c];
    }
    Vb[r][c] = f32_to_bf16(v);
  }
  __syncthreads();

  for (int t = 0; t < CLEN; ++t) {
    v8f acc = v8f_zero();
    if (mode == 0) {
      #pragma unroll
      for (int r = 0; r < 8; ++r) {
        const int m = r + 8*half;
        const int R = g*16 + m, cch = R >> 2, b = R & 3;
        const size_t row = (size_t)b*SEQ + (size_t)cch*CLEN + t;
        acc[r] = hs[row*NST + 16*w + ln];
      }
    }
    #pragma unroll
    for (int kb = 0; kb < 8; ++kb) {
      FragB a;
      a.q[0] = *(const uint4*)&Vb[ln][kb*32 + half*8];
      a.q[1] = *(const uint4*)&Vb[ln][kb*32 + 16 + half*8];
      acc = __builtin_amdgcn_wmma_f32_16x16x32_bf16(
          false, a.v, false, bf[kb].v, (short)0, acc, false, false);
    }
    __syncthreads();   // all waves done reading Vb
    #pragma unroll
    for (int r = 0; r < 8; ++r) {
      const int m = r + 8*half;
      const int R = g*16 + m, cch = R >> 2, b = R & 3;
      const size_t idx = ((size_t)b*SEQ + (size_t)cch*CLEN + t)*NST + 16*w + ln;
      const float vnew = acc[r];
      hs[idx] = (mode == 1) ? (hs[idx] + vnew) : vnew;
      Vb[m][16*w + ln] = f32_to_bf16(vnew);
    }
    __syncthreads();   // Vb updated for next step
  }
}

// ---------------------------------------------------------------------------
// Pass 2: sequential carry over chunks (single block, 16 waves).
// carryIn[0]=0; carryIn[c] = carryIn[c-1]@A^L + E_{c-1},
// E_{c-1} = hs[b*SEQ + c*CLEN - 1]. Rows 4..15 stay zero.
// ---------------------------------------------------------------------------
__global__ __launch_bounds__(512) void carry_kernel(
    const unsigned short* __restrict__ ALt,  // [256][256] bf16 = (A^CLEN)^T
    const float* __restrict__ hs,
    float* __restrict__ Hbuf)                // [CHUNKS][4][256]
{
  __shared__ __attribute__((aligned(16))) unsigned short Vb[16][264];
  const int tid  = threadIdx.x;
  const int w    = tid >> 5, lane = tid & 31;
  const int half = lane >> 4, ln = lane & 15;

  FragB bf[8];
  {
    const unsigned short* base = ALt + ((size_t)(16*w + ln))*NST + half*8;
    #pragma unroll
    for (int kb = 0; kb < 8; ++kb) {
      bf[kb].q[0] = *(const uint4*)(base + kb*32);
      bf[kb].q[1] = *(const uint4*)(base + kb*32 + 16);
    }
  }

  for (int i = tid; i < 4*NST; i += 512) Hbuf[i] = 0.f;   // carryIn[0] = 0
  for (int i = tid; i < 16*NST; i += 512) Vb[i >> 8][i & 255] = 0;
  __syncthreads();

  for (int c = 1; c < CHUNKS; ++c) {
    v8f acc = v8f_zero();
    #pragma unroll
    for (int r = 0; r < 8; ++r) {
      const int m = r + 8*half;
      if (m < 4) {
        const size_t row = (size_t)m*SEQ + (size_t)c*CLEN - 1;
        acc[r] = hs[row*NST + 16*w + ln];
      }
    }
    #pragma unroll
    for (int kb = 0; kb < 8; ++kb) {
      FragB a;
      a.q[0] = *(const uint4*)&Vb[ln][kb*32 + half*8];
      a.q[1] = *(const uint4*)&Vb[ln][kb*32 + 16 + half*8];
      acc = __builtin_amdgcn_wmma_f32_16x16x32_bf16(
          false, a.v, false, bf[kb].v, (short)0, acc, false, false);
    }
    __syncthreads();
    #pragma unroll
    for (int r = 0; r < 8; ++r) {
      const int m = r + 8*half;
      Vb[m][16*w + ln] = f32_to_bf16(acc[r]);
      if (m < 4) Hbuf[((size_t)c*4 + m)*NST + 16*w + ln] = acc[r];
    }
    __syncthreads();
  }
}

// ---------------------------------------------------------------------------
extern "C" void kernel_launch(void* const* d_in, const int* in_sizes, int n_in,
                              void* d_out, int out_size, void* d_ws, size_t ws_size,
                              hipStream_t stream) {
  const float* x    = (const float*)d_in[0];   // [4,4096,1024]
  const float* A    = (const float*)d_in[1];   // [256,256]
  const float* Bm   = (const float*)d_in[2];   // [256,256]
  const float* C    = (const float*)d_in[3];   // [1024,256]
  const float* D    = (const float*)d_in[4];   // [1024]
  const float* W_in = (const float*)d_in[5];   // [256,1024]
  const float* b_in = (const float*)d_in[6];   // [256]

  char* ws = (char*)d_ws;
  size_t off = 0;
  auto take = [&](size_t bytes) { char* p = ws + off; off = (off + bytes + 255) & ~(size_t)255; return p; };

  float*          uB   = (float*)         take((size_t)TROWS*NST*4);   // uB -> hs (in place)
  unsigned short* WBb  = (unsigned short*)take((size_t)NST*HID*2);
  float*          bB   = (float*)         take(NST*4);
  unsigned short* Cb   = (unsigned short*)take((size_t)HID*NST*2);
  unsigned short* At   = (unsigned short*)take((size_t)NST*NST*2);
  unsigned short* ALt  = (unsigned short*)take((size_t)NST*NST*2);
  float*          sq0  = (float*)         take((size_t)NST*NST*4);
  float*          sq1  = (float*)         take((size_t)NST*NST*4);
  float*          Hbuf = (float*)         take((size_t)CHUNKS*4*NST*4);
  (void)in_sizes; (void)n_in; (void)out_size; (void)ws_size;

  // --- weight prep (all tiny, L2-resident) ---
  prep_wb_kernel<<<NST, 256, 0, stream>>>(Bm, W_in, b_in, WBb, bB);
  transpose_to_bf16<<<(NST*NST + 255)/256, 256, 0, stream>>>(A, At, NST, NST);
  conv_bf16<<<(HID*NST + 255)/256, 256, 0, stream>>>(C, Cb, HID*NST);
  // A^128 by 7 squarings (CLEN = 128 = 2^7)
  matsq<<<NST, 256, 0, stream>>>(A,   sq0);  // A^2
  matsq<<<NST, 256, 0, stream>>>(sq0, sq1);  // A^4
  matsq<<<NST, 256, 0, stream>>>(sq1, sq0);  // A^8
  matsq<<<NST, 256, 0, stream>>>(sq0, sq1);  // A^16
  matsq<<<NST, 256, 0, stream>>>(sq1, sq0);  // A^32
  matsq<<<NST, 256, 0, stream>>>(sq0, sq1);  // A^64
  matsq<<<NST, 256, 0, stream>>>(sq1, sq0);  // A^128
  transpose_to_bf16<<<(NST*NST + 255)/256, 256, 0, stream>>>(sq0, ALt, NST, NST);

  // --- GEMM 1: uB = x @ WB^T + bB   [16384,1024]x[1024,256] ---
  gemm_bf16_kernel<<<dim3(TROWS/64, NST/128), 256, 0, stream>>>(
      x, WBb, bB, uB, TROWS, HID, NST);

  // --- recurrence: chunked parallel prefix scan (in place uB -> hs) ---
  scan_kernel <<<8, 512, 0, stream>>>(At, uB, Hbuf, 0);   // local scan
  carry_kernel<<<1, 512, 0, stream>>>(ALt, uB, Hbuf);     // chunk carries
  scan_kernel <<<8, 512, 0, stream>>>(At, uB, Hbuf, 1);   // apply carries

  // --- GEMM 2: y = hs @ C^T + D   [16384,256]x[256,1024] ---
  gemm_bf16_kernel<<<dim3(TROWS/64, HID/128), 256, 0, stream>>>(
      uB, Cb, D, (float*)d_out, TROWS, NST, HID);
}
